// GINGrapherLayer_44590350467907
// MI455X (gfx1250) — matrix operand order repaired and checked by reading, not compiled, not run on hardware
//
#include <hip/hip_runtime.h>

// ---------------------------------------------------------------------------
// Types for CDNA5 WMMA (wave32, v_wmma_f32_16x16x32_bf16)
// ---------------------------------------------------------------------------
typedef __bf16 bf16x4  __attribute__((ext_vector_type(4)));
typedef __bf16 bf16x8  __attribute__((ext_vector_type(8)));
typedef __bf16 bf16x16 __attribute__((ext_vector_type(16)));
typedef float  f32x8   __attribute__((ext_vector_type(8)));

union Frag {
    bf16x16 v;
    bf16x8  h[2];
};

#define DHID 128

// ---------------------------------------------------------------------------
// fp32 -> bf16 cast, 8 elements / thread (n must be a multiple of 8)
// ---------------------------------------------------------------------------
__global__ void k_cast_bf16(const float* __restrict__ src, __bf16* __restrict__ dst, int n) {
    int i = (blockIdx.x * blockDim.x + threadIdx.x) * 8;
    if (i >= n) return;
    float4 a = *(const float4*)(src + i);
    float4 b = *(const float4*)(src + i + 4);
    bf16x8 o;
    o[0] = (__bf16)a.x; o[1] = (__bf16)a.y; o[2] = (__bf16)a.z; o[3] = (__bf16)a.w;
    o[4] = (__bf16)b.x; o[5] = (__bf16)b.y; o[6] = (__bf16)b.z; o[7] = (__bf16)b.w;
    *(bf16x8*)(dst + i) = o;
}

// dstF = src (fp32), dstB = bf16(src); 4 elements / thread
__global__ void k_cast_copy(const float* __restrict__ src, float* __restrict__ dstF,
                            __bf16* __restrict__ dstB, int n) {
    int i = (blockIdx.x * blockDim.x + threadIdx.x) * 4;
    if (i >= n) return;
    float4 v = *(const float4*)(src + i);
    *(float4*)(dstF + i) = v;
    bf16x4 o;
    o[0] = (__bf16)v.x; o[1] = (__bf16)v.y; o[2] = (__bf16)v.z; o[3] = (__bf16)v.w;
    *(bf16x4*)(dstB + i) = o;
}

// outF = a + b (fp32) ; outB = bf16(a + b); 4 elements / thread
__global__ void k_add_cast(const float* __restrict__ a, const float* __restrict__ b,
                           float* __restrict__ outF, __bf16* __restrict__ outB, int n) {
    int i = (blockIdx.x * blockDim.x + threadIdx.x) * 4;
    if (i >= n) return;
    float4 va = *(const float4*)(a + i);
    float4 vb = *(const float4*)(b + i);
    float4 v = make_float4(va.x + vb.x, va.y + vb.y, va.z + vb.z, va.w + vb.w);
    *(float4*)(outF + i) = v;
    bf16x4 o;
    o[0] = (__bf16)v.x; o[1] = (__bf16)v.y; o[2] = (__bf16)v.z; o[3] = (__bf16)v.w;
    *(bf16x4*)(outB + i) = o;
}

// ---------------------------------------------------------------------------
// GEMM:  C[N,M] = A[N,K](bf16) @ B[K,M](bf16) + bias[M]   (f32 accumulate)
// Block tile 128x64, 8 waves. Wave w -> rows [16w,16w+16), 4 subtiles of 16
// cols. K staged through LDS (chunks of 32, +8 bf16 pad). If stats != null the
// epilogue also reduces per-column sum/sumsq (BN batch statistics) via LDS and
// 2 atomics/column/block — saves a full re-read of C for BN.
// Fragment layouts per CDNA5 ISA 7.12.2 (wave32).
// ---------------------------------------------------------------------------
#define LDA_PAD 40
#define LDB_PAD 40

__global__ __launch_bounds__(256)
void k_gemm_bias(const __bf16* __restrict__ A, const __bf16* __restrict__ B,
                 const float* __restrict__ bias, float* __restrict__ C,
                 float* __restrict__ stats, int Nr, int K, int M) {
    __shared__ __bf16 sA[128 * LDA_PAD];
    __shared__ __bf16 sB[64 * LDB_PAD];
    __shared__ float  red[2][16][64];   // [stat][wave*2+half][local col]

    const int tid  = threadIdx.x;
    const int wave = tid >> 5;
    const int lane = tid & 31;
    const int row0 = blockIdx.x * 128;
    const int col0 = blockIdx.y * 64;

    f32x8 acc[4] = {};

    const int r16  = lane & 15;
    const int half = lane >> 4;

    for (int k0 = 0; k0 < K; k0 += 32) {
        // ---- stage A tile: 128 rows x 32 k (512 chunks of 8 bf16; 2/thread)
        #pragma unroll
        for (int i = 0; i < 2; ++i) {
            int c  = tid + 256 * i;
            int r  = c >> 2;
            int kk = (c & 3) * 8;
            int gr = row0 + r;
            bf16x8 v = {};
            if (gr < Nr) {
                v = *(const bf16x8*)(A + (size_t)gr * K + k0 + kk);
                if (k0 + 32 < K)  // prefetch next K-chunk (global_prefetch_b8)
                    __builtin_prefetch(A + (size_t)gr * K + k0 + 32 + kk, 0, 0);
            }
            *(bf16x8*)(&sA[r * LDA_PAD + kk]) = v;
        }
        // ---- stage B tile transposed: sB[n][k] = B[k0+k][col0+n] (64x32)
        {
            int n  = tid & 63;
            int kk = (tid >> 6) * 8;
            #pragma unroll
            for (int j = 0; j < 8; ++j)
                sB[n * LDB_PAD + kk + j] = B[(size_t)(k0 + kk + j) * M + col0 + n];
        }
        __syncthreads();

        // ---- A fragment: row = wave*16 + (lane&15); K = half*8+{0..7}, +16
        Frag af;
        af.h[0] = *(const bf16x8*)(&sA[(wave * 16 + r16) * LDA_PAD + half * 8]);
        af.h[1] = *(const bf16x8*)(&sA[(wave * 16 + r16) * LDA_PAD + half * 8 + 16]);

        #pragma unroll
        for (int s = 0; s < 4; ++s) {
            // B fragment: col = s*16 + (lane&15); K = half*8+{0..7}, +16
            Frag bf;
            const __bf16* bp = &sB[(s * 16 + r16) * LDB_PAD + half * 8];
            bf.h[0] = *(const bf16x8*)(bp);
            bf.h[1] = *(const bf16x8*)(bp + 16);
            acc[s] = __builtin_amdgcn_wmma_f32_16x16x32_bf16(
                false, af.v, false, bf.v, (short)0, acc[s], false, false);
        }
        __syncthreads();
    }

    // ---- epilogue: bias + store + (optional) fused BN column statistics
    // C/D layout: row = i + 8*half, col = lane&15
    const int cw = wave * 2 + half;   // 0..15
    #pragma unroll
    for (int s = 0; s < 4; ++s) {
        int lc = s * 16 + r16;
        int gc = col0 + lc;
        float bv = bias[gc];
        float cs = 0.f, cq = 0.f;
        #pragma unroll
        for (int i = 0; i < 8; ++i) {
            int gr = row0 + wave * 16 + half * 8 + i;
            if (gr < Nr) {
                float v = acc[s][i] + bv;
                C[(size_t)gr * M + gc] = v;
                cs += v;
                cq += v * v;
            }
        }
        red[0][cw][lc] = cs;
        red[1][cw][lc] = cq;
    }
    if (stats) {
        __syncthreads();
        if (tid < 128) {
            int st = tid >> 6;      // 0 = sum, 1 = sumsq
            int c  = tid & 63;
            float t = 0.f;
            #pragma unroll
            for (int j = 0; j < 16; ++j) t += red[st][j][c];
            atomicAdd(&stats[st * M + col0 + c], t);
        }
    }
}

// ---------------------------------------------------------------------------
// BN epilogue: v = g*(y-mean)*rsqrt(var+eps)+b ; optional exact GELU; optional
// residual; writes fp32 (outF) and/or bf16 (outB). 4 elements/thread; M is a
// power of two (>=128) so the 4 columns are consecutive.
// ---------------------------------------------------------------------------
__global__ void k_bn_ep(const float* __restrict__ Y, const float* __restrict__ stats,
                        const float* __restrict__ g, const float* __restrict__ b,
                        const float* __restrict__ resid,
                        float* __restrict__ outF, __bf16* __restrict__ outB,
                        int Nr, int M, float invN, int doGelu) {
    size_t i = ((size_t)blockIdx.x * blockDim.x + threadIdx.x) * 4;
    if (i >= (size_t)Nr * M) return;
    int col = (int)(i & (size_t)(M - 1));
    float4 y = *(const float4*)(Y + i);
    float vs[4] = {y.x, y.y, y.z, y.w};
    #pragma unroll
    for (int j = 0; j < 4; ++j) {
        int c = col + j;
        float mean = stats[c] * invN;
        float var  = stats[M + c] * invN - mean * mean;
        float v = g[c] * (vs[j] - mean) * rsqrtf(var + 1e-5f) + b[c];
        if (doGelu) v = 0.5f * v * (1.f + erff(v * 0.70710678118654752f));
        vs[j] = v;
    }
    if (resid) {
        float4 r = *(const float4*)(resid + i);
        vs[0] += r.x; vs[1] += r.y; vs[2] += r.z; vs[3] += r.w;
    }
    if (outF) *(float4*)(outF + i) = make_float4(vs[0], vs[1], vs[2], vs[3]);
    if (outB) {
        bf16x4 o;
        o[0] = (__bf16)vs[0]; o[1] = (__bf16)vs[1];
        o[2] = (__bf16)vs[2]; o[3] = (__bf16)vs[3];
        *(bf16x4*)(outB + i) = o;
    }
}

// ---------------------------------------------------------------------------
// GIN aggregation: AGG[dst[e]] += H[src[e]]  (32 threads/edge, float4 gather,
// 4x global_atomic_add_f32 scatter)
// ---------------------------------------------------------------------------
__global__ void k_scatter_add(const float* __restrict__ H, const int* __restrict__ src,
                              const int* __restrict__ dst, float* __restrict__ AGG, int E) {
    int t = blockIdx.x * blockDim.x + threadIdx.x;
    int e = t >> 5;
    if (e >= E) return;
    int f = (t & 31) * 4;
    int s = src[e], d = dst[e];
    float4 v = *(const float4*)(H + (size_t)s * DHID + f);
    float* p = AGG + (size_t)d * DHID + f;
    atomicAdd(p + 0, v.x);
    atomicAdd(p + 1, v.y);
    atomicAdd(p + 2, v.z);
    atomicAdd(p + 3, v.w);
}

// ---------------------------------------------------------------------------
// Orchestration
// ---------------------------------------------------------------------------
extern "C" void kernel_launch(void* const* d_in, const int* in_sizes, int n_in,
                              void* d_out, int out_size, void* d_ws, size_t ws_size,
                              hipStream_t stream) {
    const float* x      = (const float*)d_in[0];
    const int*   edge   = (const int*)d_in[1];
    const float* enc_W  = (const float*)d_in[2];
    const float* enc_b  = (const float*)d_in[3];
    const float* bn_g   = (const float*)d_in[4];
    const float* bn_b   = (const float*)d_in[5];
    const float* gW1    = (const float*)d_in[6];
    const float* gb1    = (const float*)d_in[7];
    const float* gg     = (const float*)d_in[8];
    const float* gbb    = (const float*)d_in[9];
    const float* gW2    = (const float*)d_in[10];
    const float* gb2    = (const float*)d_in[11];
    const float* f2W1   = (const float*)d_in[12];
    const float* f2b1   = (const float*)d_in[13];
    const float* f2g1   = (const float*)d_in[14];
    const float* f2bb1  = (const float*)d_in[15];
    const float* f2W2   = (const float*)d_in[16];
    const float* f2b2   = (const float*)d_in[17];
    const float* f2g2   = (const float*)d_in[18];
    const float* f2bb2  = (const float*)d_in[19];
    const float* fnW1   = (const float*)d_in[20];
    const float* fnb1   = (const float*)d_in[21];
    const float* fng1   = (const float*)d_in[22];
    const float* fnbb1  = (const float*)d_in[23];
    const float* fnW2   = (const float*)d_in[24];
    const float* fnb2   = (const float*)d_in[25];
    const float* fng2   = (const float*)d_in[26];
    const float* fnbb2  = (const float*)d_in[27];

    const int N = in_sizes[0] / DHID;           // 50000
    const int E = in_sizes[1] / 2;              // 640000
    const int* srcI = edge;
    const int* dstI = edge + E;
    const float invN = 1.0f / (float)N;

    // ---- workspace layout (bytes) ----
    char* w = (char*)d_ws;
    const size_t OFF_WB    = 0;                               // bf16 weights
    const size_t OFF_ABF   = 491520;                          // bf16 act, N*512
    const size_t OFF_Y     = OFF_ABF + (size_t)N * 512 * 2;   // f32, N*512
    const size_t OFF_H     = OFF_Y   + (size_t)N * 512 * 4;   // f32, N*128
    const size_t OFF_R     = OFF_H   + (size_t)N * 128 * 4;   // f32, N*128 (also AGG)
    const size_t OFF_STATS = OFF_R   + (size_t)N * 128 * 4;   // f32, 2*512

    __bf16* WB    = (__bf16*)(w + OFF_WB);
    __bf16* ABF   = (__bf16*)(w + OFF_ABF);
    float*  Y     = (float*)(w + OFF_Y);
    float*  H     = (float*)(w + OFF_H);
    float*  R     = (float*)(w + OFF_R);
    float*  stats = (float*)(w + OFF_STATS);
    float*  out   = (float*)d_out;

    __bf16* wEnc = WB + 0;
    __bf16* wG1  = WB + 16384;
    __bf16* wG2  = WB + 32768;
    __bf16* wF21 = WB + 49152;   // 128x256
    __bf16* wF22 = WB + 81920;   // 256x128
    __bf16* wFn1 = WB + 114688;  // 128x512
    __bf16* wFn2 = WB + 180224;  // 512x128

    const int T = 256;
    const int rowTiles = (N + 127) / 128;
    const int nND = N * DHID;

    auto gemm = [&](const __bf16* A, const __bf16* B, const float* bias, float* C,
                    float* st, int K, int M) {
        if (st) hipMemsetAsync(st, 0, (size_t)2 * M * 4, stream);
        dim3 grid(rowTiles, M / 64);
        k_gemm_bias<<<grid, T, 0, stream>>>(A, B, bias, C, st, N, K, M);
    };
    auto bn = [&](const float* Yp, const float* g, const float* b, const float* res,
                  float* oF, __bf16* oB, int M, int gelu) {
        size_t total = (size_t)N * M / 4;
        k_bn_ep<<<(unsigned)((total + T - 1) / T), T, 0, stream>>>(
            Yp, stats, g, b, res, oF, oB, N, M, invN, gelu);
    };
    auto castN = [&](const float* s, __bf16* d, int n) {
        k_cast_bf16<<<(n / 8 + T - 1) / T, T, 0, stream>>>(s, d, n);
    };

    // ---- cast weights to bf16 ----
    castN(enc_W, wEnc, 16384);
    castN(gW1,  wG1,  16384);
    castN(gW2,  wG2,  16384);
    castN(f2W1, wF21, 32768);
    castN(f2W2, wF22, 32768);
    castN(fnW1, wFn1, 65536);
    castN(fnW2, wFn2, 65536);

    // ---- Stage 1: h = BN(x @ enc_W + enc_b) ----
    castN(x, ABF, nND);
    gemm(ABF, wEnc, enc_b, Y, stats, 128, 128);
    bn(Y, bn_g, bn_b, nullptr, H, nullptr, 128, 0);

    // ---- Stage 2: agg = h + segment_sum(h[src] -> dst)  (eps=0) ----
    hipMemcpyAsync(R, H, (size_t)nND * 4, hipMemcpyDeviceToDevice, stream); // self term
    k_scatter_add<<<(E * 32 + T - 1) / T, T, 0, stream>>>(H, srcI, dstI, R, E);
    castN(R, ABF, nND);

    // ---- Stage 3: GIN MLP ----
    gemm(ABF, wG1, gb1, Y, stats, 128, 128);
    bn(Y, gg, gbb, nullptr, nullptr, ABF, 128, 1);               // GELU -> bf16
    gemm(ABF, wG2, gb2, Y, nullptr, 128, 128);                   // no BN after
    k_cast_copy<<<(nND / 4 + T - 1) / T, T, 0, stream>>>(Y, R, ABF, nND); // R = h_gin

    // ---- Stage 4: fc2 FFN (hidden 2D) ----
    gemm(ABF, wF21, f2b1, Y, stats, 128, 256);
    bn(Y, f2g1, f2bb1, nullptr, nullptr, ABF, 256, 1);
    gemm(ABF, wF22, f2b2, Y, stats, 256, 128);
    bn(Y, f2g2, f2bb2, R, H, nullptr, 128, 0);                   // + internal residual

    // ---- Stage 5: outer residual: h = x + h_fc2 ; R := h ; ABF := bf16(h) ----
    k_add_cast<<<(nND / 4 + T - 1) / T, T, 0, stream>>>(H, x, R, ABF, nND);

    // ---- Stage 6: final FFN (hidden 4D) ----
    gemm(ABF, wFn1, fnb1, Y, stats, 128, 512);
    bn(Y, fng1, fnbb1, nullptr, nullptr, ABF, 512, 1);
    gemm(ABF, wFn2, fnb2, Y, stats, 512, 128);
    bn(Y, fng2, fnbb2, R, out, nullptr, 128, 0);                 // final output fp32
}
